// ASAPoolingModel_79559974191353
// MI455X (gfx1250) — matrix operand order repaired and checked by reading, not compiled, not run on hardware
//
#include <hip/hip_runtime.h>
#include <math.h>

typedef __attribute__((ext_vector_type(16))) _Float16 v16h;
typedef __attribute__((ext_vector_type(8)))  _Float16 v8h;
typedef __attribute__((ext_vector_type(4)))  _Float16 v4h;
typedef __attribute__((ext_vector_type(8)))  float    v8f;
typedef __attribute__((ext_vector_type(4)))  float    v4f;

constexpr int NN  = 24000;          // nodes
constexpr int EE  = 192000;         // raw edges
constexpr int EL  = EE + NN;        // edges incl. self loops = 216000
constexpr int HCc = 256;            // HID*HEADS
constexpr int HD  = 64;             // HID
constexpr int NH  = 4;              // HEADS
constexpr int KK1 = 240;            // ceil(0.01*N)
constexpr int KK2 = 2;              // ceil(0.005*K1)
constexpr float NEGS = 0.2f;

static unsigned cdivu(long long a, long long b) { return (unsigned)((a + b - 1) / b); }

// ---------------------------------------------------------------- helpers
__device__ __forceinline__ void edge_sd(const int* ei, int e, int& s, int& d) {
  if (e < EE) { s = ei[e]; d = ei[EE + e]; }
  else        { s = e - EE; d = e - EE; }
}
__device__ __forceinline__ float leaky(float x) { return x > 0.f ? x : NEGS * x; }
__device__ __forceinline__ float gelu_f(float x) {
  return 0.5f * x * (1.f + erff(x * 0.70710678118654752f));
}
__device__ __forceinline__ void atomicMaxF(float* a, float v) {
  if (v >= 0.f) atomicMax((int*)a, __float_as_int(v));
  else          atomicMin((unsigned int*)a, __float_as_uint(v));
}

// ---------------------------------------------------------------- WMMA GEMM
// C[M x Nc] = A[M x K] @ B[K x Nc], f32 row-major (logical A[m][k] =
// A[k*lda + m] when TRANSA). Block = 64 threads (2 waves) -> 32x32 C tile;
// wave w covers rows [tm + 16w, +16) with two 16x16 accumulators (2 WMMAs per
// k-step). Tiles are staged into LDS *in WMMA fragment order* so each lane
// fetches its whole fragment as two contiguous ds_load_b128. Interior tiles
// use a branch-free float4 staging path.
template <bool TRANSA>
__global__ void k_wmma_gemm_t(const float* __restrict__ A, int lda,
                              const float* __restrict__ B,
                              float* __restrict__ C, int M, int Nc, int K) {
  __shared__ __align__(32) _Float16 As[1024]; // 2 row-groups x (32 lanes x 16 halves)
  __shared__ __align__(32) _Float16 Bs[1024]; // 2 col-groups x (32 lanes x 16 halves)
  const int tid  = threadIdx.x;
  const int lane = tid & 31;
  const int w    = tid >> 5;
  const int tm = blockIdx.x * 32;
  const int tn = blockIdx.y * 32;
  const bool mnfull = (tm + 32 <= M) && (tn + 32 <= Nc);
  const bool vecok  = !TRANSA && ((lda & 3) == 0) && ((Nc & 3) == 0);
  v8f acc0 = {}, acc1 = {};
  for (int k0 = 0; k0 < K; k0 += 32) {
    if (vecok && mnfull && (k0 + 32 <= K)) {
      // ---- fast path: float4 coalesced loads, packed f16 stores ----
#pragma unroll
      for (int ch = 0; ch < 4; ++ch) {
        int li = ch * 256 + tid * 4;
        int r = li >> 5, c = li & 31;           // A: r = tile row, c = tile k
        v4f va = *(const v4f*)&A[(size_t)(tm + r) * lda + (k0 + c)];
        v4h ha = __builtin_convertvector(va, v4h);
        int g  = r >> 4, m = r & 15;
        int hi = (c >> 3) & 1;
        int fi = (c & 7) + ((c & 16) >> 1);     // 4 consecutive fragment halves
        *(v4h*)&As[g * 512 + (m + hi * 16) * 16 + fi] = ha;
      }
#pragma unroll
      for (int ch = 0; ch < 4; ++ch) {
        int li = ch * 256 + tid * 4;
        int r = li >> 5, c = li & 31;           // B: r = tile k, c = tile col
        v4f vb = *(const v4f*)&B[(size_t)(k0 + r) * Nc + (tn + c)];
        v4h hb = __builtin_convertvector(vb, v4h);
        int fi = (r & 7) + ((r & 16) >> 1), lo = ((r >> 3) & 1) * 16;
#pragma unroll
        for (int j = 0; j < 4; ++j) {
          int n = c + j;
          Bs[(n >> 4) * 512 + ((n & 15) + lo) * 16 + fi] = hb[j];
        }
      }
    } else {
      // ---- guarded scalar path (tile edges / transposed A) ----
      for (int e = 0; e < 1024; e += 64) {
        int idx = e + tid;
        int r = idx >> 5, c = idx & 31;
        int gm = tm + r, gk = k0 + c;
        float v = 0.f;
        if (gm < M && gk < K)
          v = TRANSA ? A[(size_t)gk * lda + gm] : A[(size_t)gm * lda + gk];
        int g  = r >> 4, m = r & 15;
        int hi = (c >> 3) & 1;
        int fi = (c & 7) + ((c & 16) >> 1);
        As[g * 512 + (m + hi * 16) * 16 + fi] = (_Float16)v;
      }
      for (int e = 0; e < 1024; e += 64) {
        int idx = e + tid;
        int r = idx >> 5, c = idx & 31;
        int gk = k0 + r, gn = tn + c;
        float v = 0.f;
        if (gk < K && gn < Nc) v = B[(size_t)gk * Nc + gn];
        int g = c >> 4, n = c & 15;
        Bs[g * 512 + (n + ((r >> 3) & 1) * 16) * 16 + (r & 7) + ((r & 16) >> 1)] = (_Float16)v;
      }
    }
    if (k0 + 32 < K)                            // speculative next-tile prefetch
      __builtin_prefetch(&B[(size_t)(k0 + 32 + lane) * Nc + tn], 0, 1);
    __syncthreads();
    // ---- fragment loads: contiguous 2 x ds_load_b128 per operand ----
    const _Float16* ap = &As[w * 512 + lane * 16];
    v8h a0 = *(const v8h*)ap;
    v8h a1 = *(const v8h*)(ap + 8);
    v16h af = __builtin_shufflevector(a0, a1, 0,1,2,3,4,5,6,7,8,9,10,11,12,13,14,15);
    const _Float16* bp0 = &Bs[lane * 16];
    v8h b00 = *(const v8h*)bp0;
    v8h b01 = *(const v8h*)(bp0 + 8);
    v16h bf0 = __builtin_shufflevector(b00, b01, 0,1,2,3,4,5,6,7,8,9,10,11,12,13,14,15);
    const _Float16* bp1 = &Bs[512 + lane * 16];
    v8h b10 = *(const v8h*)bp1;
    v8h b11 = *(const v8h*)(bp1 + 8);
    v16h bf1 = __builtin_shufflevector(b10, b11, 0,1,2,3,4,5,6,7,8,9,10,11,12,13,14,15);
    acc0 = __builtin_amdgcn_wmma_f32_16x16x32_f16(false, af, false, bf0,
                                                  (short)0, acc0, false, false);
    acc1 = __builtin_amdgcn_wmma_f32_16x16x32_f16(false, af, false, bf1,
                                                  (short)0, acc1, false, false);
    __syncthreads();
  }
  const int n = lane & 15;
  const int moff = (lane < 16) ? 0 : 8;
#pragma unroll
  for (int r = 0; r < 8; ++r) {
    int gm = tm + w * 16 + moff + r;
    if (gm < M) {
      int gn0 = tn + n, gn1 = tn + 16 + n;
      if (gn0 < Nc) C[(size_t)gm * Nc + gn0] = acc0[r];
      if (gn1 < Nc) C[(size_t)gm * Nc + gn1] = acc1[r];
    }
  }
}

// ---------------------------------------------------------------- elementwise
__global__ void k_fill(float* p, float v, long long n) {
  long long i = (long long)blockIdx.x * blockDim.x + threadIdx.x;
  if (i < n) p[i] = v;
}
__global__ void k_ifill(int* p, int v, long long n) {
  long long i = (long long)blockIdx.x * blockDim.x + threadIdx.x;
  if (i < n) p[i] = v;
}
// out[i] = act(in[i] + bias[i % C]); act: 0=none, 1=gelu. bias nullable.
__global__ void k_bias_act(const float* in, const float* bias, float* out,
                           long long n, int C, int act) {
  long long i = (long long)blockIdx.x * blockDim.x + threadIdx.x;
  if (i >= n) return;
  float v = in[i];
  if (bias) v += bias[(int)(i % C)];
  if (act) v = gelu_f(v);
  out[i] = v;
}

// ---------------------------------------------------------------- GAT pieces
__global__ void k_head_dots(const float* H, const float* aws, const float* awd,
                            float* os, float* od, int n) {
  int id = blockIdx.x * blockDim.x + threadIdx.x;
  if (id >= n * NH) return;
  int node = id / NH, h = id % NH;
  const float* hp = H + (size_t)node * HCc + h * HD;
  float s = 0.f, d = 0.f;
  for (int c = 0; c < HD; ++c) { float v = hp[c]; s += v * aws[h * HD + c]; d += v * awd[h * HD + c]; }
  os[id] = s; od[id] = d;
}
__global__ void k_edge_max(const int* ei, const float* Asv, const float* Bdv,
                           const float* cb, float* m, int nH, int nE) {
  long long id = (long long)blockIdx.x * blockDim.x + threadIdx.x;
  if (id >= (long long)nE * nH) return;
  int e = (int)(id / nH), h = (int)(id % nH);
  int s, d; edge_sd(ei, e, s, d);
  float v = Asv[(size_t)s * nH + h] + Bdv[(size_t)d * nH + h] + (cb ? cb[0] : 0.f);
  atomicMaxF(&m[(size_t)d * nH + h], leaky(v));
}
__global__ void k_edge_expsum(const int* ei, const float* Asv, const float* Bdv,
                              const float* cb, const float* m, float* ex,
                              float* den, int nH, int nE) {
  long long id = (long long)blockIdx.x * blockDim.x + threadIdx.x;
  if (id >= (long long)nE * nH) return;
  int e = (int)(id / nH), h = (int)(id % nH);
  int s, d; edge_sd(ei, e, s, d);
  float v = Asv[(size_t)s * nH + h] + Bdv[(size_t)d * nH + h] + (cb ? cb[0] : 0.f);
  float ev = expf(leaky(v) - m[(size_t)d * nH + h]);
  ex[id] = ev;
  atomicAdd(&den[(size_t)d * nH + h], ev);
}
__global__ void k_edge_agg(const int* ei, const float* X, const float* ex,
                           const float* den, float* out, int C, int nH, int nE) {
  long long id = (long long)blockIdx.x * blockDim.x + threadIdx.x;
  if (id >= (long long)nE * C) return;
  int e = (int)(id / C), c = (int)(id % C);
  int s, d; edge_sd(ei, e, s, d);
  int h = c / (C / nH);
  float alpha = ex[(size_t)e * nH + h] / den[(size_t)d * nH + h];
  atomicAdd(&out[(size_t)d * C + c], X[(size_t)s * C + c] * alpha);
}
__global__ void k_edge_featmax(const int* ei, const float* X, float* xq, int C, int nE) {
  long long id = (long long)blockIdx.x * blockDim.x + threadIdx.x;
  if (id >= (long long)nE * C) return;
  int e = (int)(id / C), c = (int)(id % C);
  int s, d; edge_sd(ei, e, s, d);
  atomicMaxF(&xq[(size_t)d * C + c], X[(size_t)s * C + c]);
}
__global__ void k_edge_aggscalar(const int* ei, const float* a, const float* b,
                                 float* agg, int nE) {
  int e = blockIdx.x * blockDim.x + threadIdx.x;
  if (e >= nE) return;
  int s, d; edge_sd(ei, e, s, d);
  atomicAdd(&agg[d], a[s] - b[d]);
}

// ---------------------------------------------------------------- vec ops
__global__ void k_matvec(const float* X, const float* w, const float* b0,
                         float* y, int n, int C) {
  int gid = blockIdx.x * blockDim.x + threadIdx.x;
  int row = gid >> 5, lane = gid & 31;
  if (row >= n) return;
  float s = 0.f;
  for (int c = lane; c < C; c += 32) s += X[(size_t)row * C + c] * w[c];
#pragma unroll
  for (int o = 16; o > 0; o >>= 1) s += __shfl_down(s, o, 32);
  if (lane == 0) y[row] = s + (b0 ? b0[0] : 0.f);
}
__global__ void k_fitness(const float* X, const float* w, const float* b0,
                          const float* agg, float* y, int n, int C) {
  int gid = blockIdx.x * blockDim.x + threadIdx.x;
  int row = gid >> 5, lane = gid & 31;
  if (row >= n) return;
  float s = 0.f;
  for (int c = lane; c < C; c += 32) s += X[(size_t)row * C + c] * w[c];
#pragma unroll
  for (int o = 16; o > 0; o >>= 1) s += __shfl_down(s, o, 32);
  if (lane == 0) { float t = s + b0[0] + agg[row]; y[row] = 1.f / (1.f + expf(-t)); }
}

// ---------------------------------------------------------------- top-k
__global__ void k_topk(const float* val, int n, int K, float* scratch,
                       int* perm, float* outfit) {
  __shared__ float sv[256];
  __shared__ int   si[256];
  int t = threadIdx.x;
  for (int i = t; i < n; i += 256) scratch[i] = val[i];
  __syncthreads();
  for (int k = 0; k < K; ++k) {
    float best = -INFINITY; int bi = 0x7fffffff;
    for (int i = t; i < n; i += 256) {
      float v = scratch[i];
      if (v > best || (v == best && i < bi)) { best = v; bi = i; }
    }
    sv[t] = best; si[t] = bi;
    __syncthreads();
    for (int s = 128; s > 0; s >>= 1) {
      if (t < s) {
        if (sv[t + s] > sv[t] || (sv[t + s] == sv[t] && si[t + s] < si[t])) {
          sv[t] = sv[t + s]; si[t] = si[t + s];
        }
      }
      __syncthreads();
    }
    if (t == 0) { perm[k] = si[0]; outfit[k] = sv[0]; scratch[si[0]] = -INFINITY; }
    __syncthreads();
  }
}
__global__ void k_select_scale(const float* X, const int* perm, const float* fit,
                               float* out, int K, int C) {
  int id = blockIdx.x * blockDim.x + threadIdx.x;
  if (id >= K * C) return;
  int k = id / C, c = id % C;
  out[id] = X[(size_t)perm[k] * C + c] * fit[k];
}
__global__ void k_icolset(int* colidx, const int* perm, int K) {
  int t = threadIdx.x;
  if (t < K) colidx[perm[t]] = t;
}
__global__ void k_scatterS(const int* ei, const int* colidx, const float* ex,
                           const float* den, float* S, int nE) {
  int e = blockIdx.x * blockDim.x + threadIdx.x;
  if (e >= nE) return;
  int s, d; edge_sd(ei, e, s, d);
  int c = colidx[d];
  if (c < KK1) atomicAdd(&S[(size_t)s * KK1 + c], ex[e] / den[d]);
}
__global__ void k_scatterAS(const int* ei, const float* S, float* AS, int nE) {
  long long id = (long long)blockIdx.x * blockDim.x + threadIdx.x;
  if (id >= (long long)nE * KK1) return;
  int e = (int)(id / KK1), c = (int)(id % KK1);
  int s, d; edge_sd(ei, e, s, d);
  atomicAdd(&AS[(size_t)s * KK1 + c], S[(size_t)d * KK1 + c]);
}
__global__ void k_diag1(float* A, int kk) {
  int t = blockIdx.x * blockDim.x + threadIdx.x;
  if (t < kk) A[(size_t)t * kk + t] = 1.f;
}

// ---------------------------------------------------------------- dense blocks
__global__ void k_dense_gat(const float* H, const float* asv, const float* adv,
                            const float* A, const float* bvec, float* out, int kk) {
  __shared__ float alpha[KK1];
  __shared__ float red[256];
  int i = blockIdx.x, h = blockIdx.y, t = threadIdx.x;
  float e = -INFINITY;
  if (t < kk && A[(size_t)t * kk + i] != 0.f)
    e = leaky(asv[t * NH + h] + adv[i * NH + h]);
  red[t] = e;
  __syncthreads();
  for (int s = 128; s > 0; s >>= 1) { if (t < s) red[t] = fmaxf(red[t], red[t + s]); __syncthreads(); }
  float mx = red[0];
  __syncthreads();
  float ev = (t < kk) ? expf(e - mx) : 0.f;
  if (t < kk) alpha[t] = ev;
  red[t] = ev;
  __syncthreads();
  for (int s = 128; s > 0; s >>= 1) { if (t < s) red[t] += red[t + s]; __syncthreads(); }
  float dsum = red[0];
  __syncthreads();
  if (t < HD) {
    float acc = 0.f;
    for (int j = 0; j < kk; ++j) acc += alpha[j] * H[(size_t)j * HCc + h * HD + t];
    out[(size_t)i * HCc + h * HD + t] = acc / dsum + bvec[h * HD + t];
  }
}
__global__ void k_dense_segmax(const float* X, const float* A, float* xq, int kk, int C) {
  int i = blockIdx.x, c = threadIdx.x;
  float mx = -INFINITY;
  for (int j = 0; j < kk; ++j)
    if (A[(size_t)j * kk + i] != 0.f) mx = fmaxf(mx, X[(size_t)j * C + c]);
  xq[(size_t)i * C + c] = mx;
}
__global__ void k_dense_S(const float* sx, const float* q, const float* attb,
                          const float* A, float* St, int kk) {
  __shared__ float red[256];
  int i = blockIdx.x, t = threadIdx.x;
  float e = -INFINITY;
  if (t < kk && A[(size_t)t * kk + i] != 0.f) e = leaky(sx[t] + q[i] + attb[0]);
  red[t] = e;
  __syncthreads();
  for (int s = 128; s > 0; s >>= 1) { if (t < s) red[t] = fmaxf(red[t], red[t + s]); __syncthreads(); }
  float mx = red[0];
  __syncthreads();
  float ev = (t < kk) ? expf(e - mx) : 0.f;
  red[t] = ev;
  __syncthreads();
  for (int s = 128; s > 0; s >>= 1) { if (t < s) red[t] += red[t + s]; __syncthreads(); }
  float dsum = red[0];
  if (t < kk) St[(size_t)i * kk + t] = ev / dsum;
}
__global__ void k_colsum(const float* A, float* degv, int kk) {
  int i = blockIdx.x * blockDim.x + threadIdx.x;
  if (i >= kk) return;
  float s = 0.f;
  for (int j = 0; j < kk; ++j) s += A[(size_t)j * kk + i];
  degv[i] = s;
}
__global__ void k_dense_agg(const float* A, const float* a, const float* b,
                            const float* degv, float* agg, int kk) {
  int i = blockIdx.x * blockDim.x + threadIdx.x;
  if (i >= kk) return;
  float s = 0.f;
  for (int j = 0; j < kk; ++j) s += A[(size_t)j * kk + i] * a[j];
  agg[i] = s - degv[i] * b[i];
}
__global__ void k_colpool(const float* X, int rows, int C, float* mx, float* mean) {
  int c = threadIdx.x;
  if (c >= C) return;
  float m = -INFINITY, s = 0.f;
  for (int r = 0; r < rows; ++r) { float v = X[(size_t)r * C + c]; m = fmaxf(m, v); s += v; }
  mx[c] = m; mean[c] = s / rows;
}
__global__ void k_bias2(const float* mx1, const float* mn1, const float* mx2,
                        const float* mn2, const float* W, const float* b, float* o) {
  int t = threadIdx.x;
  if (t >= HD) return;
  float acc = b[t];
  for (int c = 0; c < HCc; ++c) {
    float gmax = mx1[c] + mx2[c];
    float gmean = mn1[c] + mn2[c];
    acc += gmax * W[(size_t)(HCc + c) * HD + t] + gmean * W[(size_t)(2 * HCc + c) * HD + t];
  }
  o[t] = acc;
}

// ---------------------------------------------------------------- driver
extern "C" void kernel_launch(void* const* d_in, const int* in_sizes, int n_in,
                              void* d_out, int out_size, void* d_ws, size_t ws_size,
                              hipStream_t stream) {
  (void)in_sizes; (void)n_in; (void)out_size; (void)ws_size;
  const float* x  = (const float*)d_in[0];
  const int*   ei = (const int*)d_in[1];   // [src(E) | dst(E)]
  int qi = 3;
  auto nf = [&]() { const float* p = (const float*)d_in[qi]; ++qi; return p; };
  const float *g1aW = nf(), *g1aS = nf(), *g1aD = nf(), *g1ab = nf();
  const float *g1bW = nf(), *g1bS = nf(), *g1bD = nf(), *g1bb = nf();
  const float *g2aW = nf(), *g2aS = nf(), *g2aD = nf(), *g2ab = nf();
  const float *g2bW = nf(), *g2bS = nf(), *g2bD = nf(), *g2bb = nf();
  const float *p1linW = nf(), *p1linb = nf(), *p1attW = nf(), *p1attb = nf(),
              *p1l1W = nf(), *p1l1b = nf(), *p1l2W = nf(), *p1l3W = nf(), *p1l3b = nf();
  const float *p2linW = nf(), *p2linb = nf(), *p2attW = nf(), *p2attb = nf(),
              *p2l1W = nf(), *p2l1b = nf(), *p2l2W = nf(), *p2l3W = nf(), *p2l3b = nf();
  const float *ph1W = nf(), *ph1b = nf(), *ph2W = nf(), *ph2b = nf();
  float* out = (float*)d_out;

  // -------- workspace carve-out (~106 MB of f32) --------
  float* wsf = (float*)d_ws;
  size_t off = 0;
  auto FA = [&](size_t n) { float* p = wsf + off; off += n; return p; };
  float* B0 = FA((size_t)NN * HCc);
  float* B1 = FA((size_t)NN * HCc);
  float* B2 = FA((size_t)NN * HCc);
  float* B3 = FA((size_t)NN * HCc);
  float* exb = FA((size_t)EL * NH);
  float* mb  = FA((size_t)NN * NH);
  float* db  = FA((size_t)NN * NH);
  float* asb = FA((size_t)NN * NH);
  float* adb = FA((size_t)NN * NH);
  float* qv   = FA(NN);
  float* sxv  = FA(NN);
  float* av   = FA(NN);
  float* bvv  = FA(NN);
  float* aggv = FA(NN);
  float* fitv = FA(NN);
  float* scr  = FA(NN);
  int*   colidx = (int*)FA(NN);
  int*   perm1  = (int*)FA(256);
  float* fit1   = FA(256);
  int*   perm2  = (int*)FA(8);
  float* fit2   = FA(8);
  float* zp1 = FA((size_t)KK1 * HCc);
  float* A1  = FA((size_t)KK1 * KK1);
  float* St  = FA((size_t)KK1 * KK1);
  float* D0  = FA((size_t)KK1 * HCc);
  float* D1  = FA((size_t)KK1 * HCc);
  float* D2  = FA((size_t)KK1 * HCc);
  float* qd   = FA(KK1);
  float* sxd  = FA(KK1);
  float* avd  = FA(KK1);
  float* bvd  = FA(KK1);
  float* degv = FA(KK1);
  float* aggd = FA(KK1);
  float* fitd = FA(KK1);
  float* zp2 = FA((size_t)KK2 * HCc);
  float* mx1 = FA(HCc); float* mn1 = FA(HCc);
  float* mx2 = FA(HCc); float* mn2 = FA(HCc);
  float* bias2 = FA(HD);

  auto gemm = [&](const float* A, int lda, const float* B, float* C,
                  int M, int Nc, int K) {
    dim3 g((M + 31) / 32, (Nc + 31) / 32);
    k_wmma_gemm_t<false><<<g, 64, 0, stream>>>(A, lda, B, C, M, Nc, K);
  };
  auto gemmT = [&](const float* A, int lda, const float* B, float* C,
                   int M, int Nc, int K) {
    dim3 g((M + 31) / 32, (Nc + 31) / 32);
    k_wmma_gemm_t<true><<<g, 64, 0, stream>>>(A, lda, B, C, M, Nc, K);
  };
  auto fillf = [&](float* p, float v, long long n) {
    k_fill<<<cdivu(n, 256), 256, 0, stream>>>(p, v, n);
  };
  auto mv = [&](const float* X, const float* w, const float* b0, float* y, int n) {
    k_matvec<<<cdivu((long long)n * 32, 256), 256, 0, stream>>>(X, w, b0, y, n, HCc);
  };

  // GAT sparse layer: O = act(aggregate(softmax-attn, X@W) + b)
  auto run_gat = [&](const float* Xin, int fin, const float* Wt, const float* aws,
                     const float* awd, const float* bv, float* Hbuf, float* Obuf, int act) {
    gemm(Xin, fin, Wt, Hbuf, NN, HCc, fin);
    k_head_dots<<<cdivu((long long)NN * NH, 256), 256, 0, stream>>>(Hbuf, aws, awd, asb, adb, NN);
    fillf(mb, -INFINITY, (long long)NN * NH);
    fillf(db, 0.f, (long long)NN * NH);
    k_edge_max<<<cdivu((long long)EL * NH, 256), 256, 0, stream>>>(ei, asb, adb, nullptr, mb, NH, EL);
    k_edge_expsum<<<cdivu((long long)EL * NH, 256), 256, 0, stream>>>(ei, asb, adb, nullptr, mb, exb, db, NH, EL);
    fillf(Obuf, 0.f, (long long)NN * HCc);
    k_edge_agg<<<cdivu((long long)EL * HCc, 256), 256, 0, stream>>>(ei, Hbuf, exb, db, Obuf, HCc, NH, EL);
    k_bias_act<<<cdivu((long long)NN * HCc, 256), 256, 0, stream>>>(Obuf, bv, Obuf, (long long)NN * HCc, HCc, act);
  };

  // ---- GAT stack ----
  run_gat(x,  32,  g1aW, g1aS, g1aD, g1ab, B2, B1, 1);       // gelu(gat g1a)
  run_gat(B1, HCc, g1bW, g1bS, g1bD, g1bb, B2, B0, 0);       // z1 = B0
  k_bias_act<<<cdivu((long long)NN * HCc, 256), 256, 0, stream>>>(B0, nullptr, B1, (long long)NN * HCc, HCc, 1); // B1 = gelu(z1)
  run_gat(B1, HCc, g2aW, g2aS, g2aD, g2ab, B2, B3, 1);       // gelu(gat g2a)
  run_gat(B3, HCc, g2bW, g2bS, g2bD, g2bb, B1, B2, 0);       // z2 = B2

  // ---- ASAP sparse on z1 = B0 ----
  fillf(B1, -INFINITY, (long long)NN * HCc);                  // xq
  k_edge_featmax<<<cdivu((long long)EL * HCc, 256), 256, 0, stream>>>(ei, B0, B1, HCc, EL);
  gemm(B1, HCc, p1linW, B3, NN, HCc, HCc);
  k_bias_act<<<cdivu((long long)NN * HCc, 256), 256, 0, stream>>>(B3, p1linb, B3, (long long)NN * HCc, HCc, 0);
  mv(B3, p1attW, nullptr, qv, NN);                            // q  = lin(xq) . attW[:C]
  mv(B0, p1attW + HCc, nullptr, sxv, NN);                     // sx = z1 . attW[C:]
  fillf(mb, -INFINITY, NN);
  fillf(db, 0.f, NN);
  k_edge_max<<<cdivu(EL, 256), 256, 0, stream>>>(ei, sxv, qv, p1attb, mb, 1, EL);
  k_edge_expsum<<<cdivu(EL, 256), 256, 0, stream>>>(ei, sxv, qv, p1attb, mb, exb, db, 1, EL);
  fillf(B1, 0.f, (long long)NN * HCc);                        // xnew
  k_edge_agg<<<cdivu((long long)EL * HCc, 256), 256, 0, stream>>>(ei, B0, exb, db, B1, HCc, 1, EL);
  mv(B1, p1l1W, p1l1b, av, NN);
  mv(B1, p1l2W, nullptr, bvv, NN);
  fillf(aggv, 0.f, NN);
  k_edge_aggscalar<<<cdivu(EL, 256), 256, 0, stream>>>(ei, av, bvv, aggv, EL);
  k_fitness<<<cdivu((long long)NN * 32, 256), 256, 0, stream>>>(B1, p1l3W, p1l3b, aggv, fitv, NN, HCc);
  k_topk<<<1, 256, 0, stream>>>(fitv, NN, KK1, scr, perm1, fit1);
  k_select_scale<<<cdivu(KK1 * HCc, 256), 256, 0, stream>>>(B1, perm1, fit1, zp1, KK1, HCc);
  k_ifill<<<cdivu(NN, 256), 256, 0, stream>>>(colidx, KK1, NN);
  k_icolset<<<1, 256, 0, stream>>>(colidx, perm1, KK1);
  float* Sm = B1;   // xnew dead -> reuse for S (N x K1)
  float* ASm = B3;  // reuse for AS (N x K1)
  fillf(Sm, 0.f, (long long)NN * KK1);
  k_scatterS<<<cdivu(EL, 256), 256, 0, stream>>>(ei, colidx, exb, db, Sm, EL);
  fillf(ASm, 0.f, (long long)NN * KK1);
  k_scatterAS<<<cdivu((long long)EL * KK1, 256), 256, 0, stream>>>(ei, Sm, ASm, EL);
  gemmT(Sm, KK1, ASm, A1, KK1, KK1, NN);                      // A2 = S^T @ AS
  k_diag1<<<1, 256, 0, stream>>>(A1, KK1);

  // ---- dense GAT x2 on zp1 ----
  gemm(zp1, HCc, g2aW, D0, KK1, HCc, HCc);
  k_head_dots<<<cdivu(KK1 * NH, 256), 256, 0, stream>>>(D0, g2aS, g2aD, asb, adb, KK1);
  k_dense_gat<<<dim3(KK1, NH), 256, 0, stream>>>(D0, asb, adb, A1, g2ab, D1, KK1);
  k_bias_act<<<cdivu((long long)KK1 * HCc, 256), 256, 0, stream>>>(D1, nullptr, D1, (long long)KK1 * HCc, HCc, 1);
  gemm(D1, HCc, g2bW, D0, KK1, HCc, HCc);
  k_head_dots<<<cdivu(KK1 * NH, 256), 256, 0, stream>>>(D0, g2bS, g2bD, asb, adb, KK1);
  k_dense_gat<<<dim3(KK1, NH), 256, 0, stream>>>(D0, asb, adb, A1, g2bb, D2, KK1);  // zp11 = D2

  // ---- ASAP dense ----
  k_dense_segmax<<<KK1, 256, 0, stream>>>(D2, A1, D0, KK1, HCc);
  gemm(D0, HCc, p2linW, D1, KK1, HCc, HCc);
  k_bias_act<<<cdivu((long long)KK1 * HCc, 256), 256, 0, stream>>>(D1, p2linb, D1, (long long)KK1 * HCc, HCc, 0);
  k_matvec<<<cdivu((long long)KK1 * 32, 256), 256, 0, stream>>>(D1, p2attW, nullptr, qd, KK1, HCc);
  k_matvec<<<cdivu((long long)KK1 * 32, 256), 256, 0, stream>>>(D2, p2attW + HCc, nullptr, sxd, KK1, HCc);
  k_dense_S<<<KK1, 256, 0, stream>>>(sxd, qd, p2attb, A1, St, KK1);
  gemm(St, KK1, D2, D0, KK1, HCc, KK1);                       // xnew = S^T @ zp11
  k_matvec<<<cdivu((long long)KK1 * 32, 256), 256, 0, stream>>>(D0, p2l1W, p2l1b, avd, KK1, HCc);
  k_matvec<<<cdivu((long long)KK1 * 32, 256), 256, 0, stream>>>(D0, p2l2W, nullptr, bvd, KK1, HCc);
  k_colsum<<<1, 256, 0, stream>>>(A1, degv, KK1);
  k_dense_agg<<<1, 256, 0, stream>>>(A1, avd, bvd, degv, aggd, KK1);
  k_fitness<<<cdivu((long long)KK1 * 32, 256), 256, 0, stream>>>(D0, p2l3W, p2l3b, aggd, fitd, KK1, HCc);
  k_topk<<<1, 256, 0, stream>>>(fitd, KK1, KK2, scr, perm2, fit2);
  k_select_scale<<<cdivu(KK2 * HCc, 256), 256, 0, stream>>>(D0, perm2, fit2, zp2, KK2, HCc);

  // ---- pooling + prediction head (rank-1 folded broadcast features) ----
  k_colpool<<<1, 256, 0, stream>>>(zp1, KK1, HCc, mx1, mn1);
  k_colpool<<<1, 256, 0, stream>>>(zp2, KK2, HCc, mx2, mn2);
  k_bias2<<<1, 64, 0, stream>>>(mx1, mn1, mx2, mn2, ph1W, ph1b, bias2);
  gemm(B2, HCc, ph1W, B1, NN, HD, HCc);                       // z2 @ ph1_W[:256]
  k_bias_act<<<cdivu((long long)NN * HD, 256), 256, 0, stream>>>(B1, bias2, B1, (long long)NN * HD, HD, 1);
  gemm(B1, HD, ph2W, out, NN, 8, HD);
  k_bias_act<<<cdivu((long long)NN * 8, 256), 256, 0, stream>>>(out, ph2b, out, (long long)NN * 8, 8, 0);
}
